// gnn_18030272709287
// MI455X (gfx1250) — compile-verified, matrix operand order checked by reading, not run on hardware
//
#include <hip/hip_runtime.h>

// ---------------------------------------------------------------------------
// EGNN-style few-shot GNN for MI455X (gfx1250, wave32, WMMA).
//
// Dominant cost: pairwise distance networks = GEMMs over M = B*N*N = 294912
// rows (K = 133/197/261/128/64, dout = 128/128/64/64).  ~95 GFLOP total.
// f16 WMMA (v_wmma_f32_16x16x32_f16) with f32 accumulation; global batch-norm
// stats accumulated atomically in each GEMM's epilogue and applied (BN+leaky)
// in the NEXT GEMM's A-load path.  Conv bias cancels exactly through BN and
// is skipped.  N-tile == dout so each A tile is staged exactly once
// (~1.2 GB total HBM traffic -> ~50 us at 23.3 TB/s); prefetch covers the
// A-stream staging latency between tile barriers.
// ---------------------------------------------------------------------------

constexpr int   Bb        = 2;
constexpr int   Nn        = 384;
constexpr int   IND       = 133;          // EMB + N_WAY
constexpr int   NWAY      = 5;
constexpr float NSHOT_INV = 1.0f / 76.0f;
constexpr int   GSTR      = 272;          // padded feature stride (floats)
constexpr int   Mtot      = Bb * Nn * Nn; // 294912 pair rows
constexpr float MINV      = 1.0f / (float)Mtot;
constexpr float BNEPS     = 1e-5f;

typedef __attribute__((ext_vector_type(16))) _Float16 v16h;
typedef __attribute__((ext_vector_type(8)))  _Float16 h8;
typedef __attribute__((ext_vector_type(8)))  float    v8f;

union FragU { v16h v; struct { h8 lo, hi; } p; };

__device__ __forceinline__ float leaky(float x) { return x >= 0.f ? x : 0.01f * x; }

__device__ __forceinline__ void atomAddShared(float* p, float v) {
  __hip_atomic_fetch_add(p, v, __ATOMIC_RELAXED, __HIP_MEMORY_SCOPE_WORKGROUP);
}
__device__ __forceinline__ void atomAddGlobal(float* p, float v) {
  __hip_atomic_fetch_add(p, v, __ATOMIC_RELAXED, __HIP_MEMORY_SCOPE_AGENT);
}

// ---------------------------------------------------------------------------
// dist_gemm<MODE, NT>:  C[64 x NT] tile of A(M x K) @ W^T(K x dout), WMMA f16.
//   MODE 0: A[m,k] = |G[b,i,k] - G[b,j,k]| built on the fly (f32 -> f16)
//   MODE 1: A[m,k] = leaky((Hin[m,k]-mean)*rstd*gamma+beta)  (f16 in)
//   NT = 64 or 128 output channels per block (NT == dout => A staged once).
// Epilogue: raw f16 output + per-channel sum / sumsq atomics for BN stats.
// Block = 128 threads = 4 waves; wave w owns rows 16w..16w+15, all NT cols.
// ---------------------------------------------------------------------------
template <int MODE, int NT>
__global__ __launch_bounds__(128)
void dist_gemm(const float* __restrict__ G, const _Float16* __restrict__ Hin,
               int K, const float* __restrict__ W,
               const float* __restrict__ fin, const float* __restrict__ gamma,
               const float* __restrict__ beta,
               _Float16* __restrict__ Hout, int dout,
               float* __restrict__ raw)
{
  constexpr int NTB    = NT / 16;                      // WMMA col tiles per wave
  constexpr int TILESZ = ((64 + NT) * 40 > 64 * NT) ? (64 + NT) * 40 : 64 * NT;
  __shared__ __align__(16) _Float16 tiles[TILESZ];     // A | B (reused as C stage)
  __shared__ float lstat[2 * NT];                      // block col sum | sumsq
  __shared__ float bnp[512];                           // mean | rstd | gamma | beta

  _Float16* tileA = tiles;
  _Float16* tileB = tiles + 64 * 40;

  const int t    = threadIdx.x;
  const int wave = t >> 5;
  const int lane = t & 31;
  const int m0   = blockIdx.x * 64;
  const int n0   = blockIdx.y * NT;

  for (int q = t; q < 2 * NT; q += 128) lstat[q] = 0.f;
  if constexpr (MODE == 1) {
    for (int k = t; k < K; k += 128) {
      bnp[k]       = fin[k];
      bnp[128 + k] = fin[K + k];
      bnp[256 + k] = gamma[k];
      bnp[384 + k] = beta[k];
    }
  }

  const float* gi  = nullptr;
  const float* gj0 = nullptr;
  if constexpr (MODE == 0) {
    // 64 | 384, so a 64-row tile has fixed (b, i) and j = j0..j0+63.
    int b   = m0 / (Nn * Nn);
    int rem = m0 - b * Nn * Nn;
    int i   = rem / Nn;
    int j0  = rem - i * Nn;
    gi  = G + (size_t)(b * Nn + i)  * GSTR;
    gj0 = G + (size_t)(b * Nn + j0) * GSTR;
  }

  const int KT = (K + 31) & ~31;
  v8f c[NTB] = {};
  const int r  = t >> 1;         // tile row (A) / first channel row (B) staged
  const int kh = (t & 1) * 16;   // which 16-wide k half

  for (int k0 = 0; k0 < KT; k0 += 32) {
    __syncthreads();
    { // ---- stage A tile: 64 rows x 32 k (f16, row stride 40) ----
      _Float16* dst = &tileA[r * 40 + kh];
      if constexpr (MODE == 0) {
        const float* pj = gj0 + (size_t)r * GSTR + k0 + kh;
        const float* pi = gi + k0 + kh;
        if (k0 + 32 < KT) __builtin_prefetch(pj + 32, 0, 1);  // global_prefetch_b8
        if (k0 + kh + 16 <= K) {
          #pragma unroll
          for (int e = 0; e < 16; e += 4) {
            float4 a4 = *(const float4*)(pj + e);
            float4 b4 = *(const float4*)(pi + e);
            dst[e + 0] = (_Float16)fabsf(a4.x - b4.x);
            dst[e + 1] = (_Float16)fabsf(a4.y - b4.y);
            dst[e + 2] = (_Float16)fabsf(a4.z - b4.z);
            dst[e + 3] = (_Float16)fabsf(a4.w - b4.w);
          }
        } else {
          #pragma unroll
          for (int e = 0; e < 16; e++) {
            int kk = k0 + kh + e;
            dst[e] = (kk < K) ? (_Float16)fabsf(pj[e] - pi[e]) : (_Float16)0.f;
          }
        }
      } else {
        const h8* src = (const h8*)(Hin + (size_t)(m0 + r) * K + k0 + kh);
        if (k0 + 32 < KT) __builtin_prefetch((const void*)(src + 4), 0, 1);
        h8 x0 = src[0];
        h8 x1 = src[1];
        #pragma unroll
        for (int e = 0; e < 8; e++) {
          int kk = k0 + kh + e;
          float f0 = (float)x0[e];
          f0 = leaky((f0 - bnp[kk]) * bnp[128 + kk] * bnp[256 + kk] + bnp[384 + kk]);
          dst[e] = (_Float16)f0;
          int kq = kk + 8;
          float f1 = (float)x1[e];
          f1 = leaky((f1 - bnp[kq]) * bnp[128 + kq] * bnp[256 + kq] + bnp[384 + kq]);
          dst[8 + e] = (_Float16)f1;
        }
      }
    }
    { // ---- stage B tile: W rows (dout,K) f32, stored [n][k] f16 ----
      #pragma unroll
      for (int nb = 0; nb < NT / 64; nb++) {
        const int n = nb * 64 + r;
        const float* wr = W + (size_t)(n0 + n) * K + k0 + kh;
        _Float16* dst = &tileB[n * 40 + kh];
        #pragma unroll
        for (int e = 0; e < 16; e++) {
          int kk = k0 + kh + e;
          dst[e] = (kk < K) ? (_Float16)wr[e] : (_Float16)0.f;
        }
      }
    }
    __syncthreads();

    // ---- fragment loads (ds_load_b128 pairs) + NTB x WMMA ----
    const int lm   = lane & 15;
    const int koff = (lane >> 4) << 3;
    FragU a;
    const _Float16* ar = &tileA[(wave * 16 + lm) * 40];
    a.p.lo = *(const h8*)(ar + koff);
    a.p.hi = *(const h8*)(ar + 16 + koff);
    #pragma unroll
    for (int tb = 0; tb < NTB; tb++) {
      FragU bfr;
      const _Float16* br = &tileB[(tb * 16 + lm) * 40];
      bfr.p.lo = *(const h8*)(br + koff);
      bfr.p.hi = *(const h8*)(br + 16 + koff);
      c[tb] = __builtin_amdgcn_wmma_f32_16x16x32_f16(
          false, a.v, false, bfr.v, (short)0, c[tb], false, false);
    }
  }

  // ---- epilogue: BN stat partials (col sum / sumsq) ----
  const int lm = lane & 15;
  #pragma unroll
  for (int tb = 0; tb < NTB; tb++) {
    float s1 = 0.f, s2 = 0.f;
    #pragma unroll
    for (int e = 0; e < 8; e++) { float v = c[tb][e]; s1 += v; s2 += v * v; }
    s1 += __shfl_xor(s1, 16, 32);   // lanes L and L^16 hold the same column
    s2 += __shfl_xor(s2, 16, 32);
    if ((lane >> 4) == 0) {
      atomAddShared(&lstat[tb * 16 + lm], s1);
      atomAddShared(&lstat[NT + tb * 16 + lm], s2);
    }
  }
  __syncthreads();

  // ---- stage C through LDS (reuse tiles) for coalesced f16 stores ----
  {
    _Float16* Ch = tiles;  // 64*NT halves <= TILESZ
    const int rowHi = (lane >> 4) << 3;
    #pragma unroll
    for (int tb = 0; tb < NTB; tb++)
      #pragma unroll
      for (int e = 0; e < 8; e++)
        Ch[(wave * 16 + e + rowHi) * NT + tb * 16 + lm] = (_Float16)c[tb][e];
  }
  __syncthreads();
  {
    const int rr  = t >> 1;
    const int ch0 = (t & 1) * (NT / 2);
    const _Float16* csrc = &tiles[rr * NT + ch0];
    _Float16* gdst = Hout + (size_t)(m0 + rr) * dout + n0 + ch0;
    #pragma unroll
    for (int e = 0; e < NT / 16; e++)
      *(uint4*)(gdst + e * 8) = *(const uint4*)(csrc + e * 8);
  }
  // ---- flush stats: one global atomic per column per block ----
  for (int q = t; q < 2 * NT; q += 128) {
    int dstIdx = (q < NT) ? (n0 + q) : (dout + n0 + (q - NT));
    atomAddGlobal(&raw[dstIdx], lstat[q]);
  }
}

// ---------------------------------------------------------------------------
// finalize BN stats: mean / rsqrt(var+eps); re-zero raw accumulators.
// ---------------------------------------------------------------------------
__global__ __launch_bounds__(256)
void finalize_stats(float* __restrict__ raw, float* __restrict__ fin, int dout)
{
  int t = threadIdx.x;
  float s = 0.f, q = 0.f;
  if (t < dout) { s = raw[t]; q = raw[dout + t]; }
  __syncthreads();
  if (t < dout) {
    float m = s * MINV;
    float v = q * MINV - m * m;
    fin[t] = m;
    fin[dout + t] = rsqrtf(v + BNEPS);
  }
  raw[t] = 0.f;
}

__global__ void zero_stats(float* __restrict__ raw) { raw[threadIdx.x] = 0.f; }

// ---------------------------------------------------------------------------
// anchor: anc[c,d] = sum_{n>=1} lab[n,c]*feat[n,d]/76; mix 0.5*proj + 0.5*feat.
// One block per batch.  Output stride GSTR.
// ---------------------------------------------------------------------------
__global__ __launch_bounds__(256)
void anchor_kernel(const float* __restrict__ Gin, int in_stride, int din,
                   float* __restrict__ Gout)
{
  __shared__ float anc[NWAY * 256];
  int t = threadIdx.x;
  int b = blockIdx.x;
  const float* base = Gin + (size_t)b * Nn * in_stride;
  const int fd = din - NWAY;

  for (int idx = t; idx < NWAY * fd; idx += 256) {
    int c = idx / fd, d = idx - c * fd;
    float acc = 0.f;
    for (int n = 1; n < Nn; n++) {
      const float* row = base + (size_t)n * in_stride;
      acc += row[fd + c] * row[d];
    }
    anc[c * 256 + d] = acc * NSHOT_INV;
  }
  __syncthreads();

  float* ob = Gout + (size_t)b * Nn * GSTR;
  for (int idx = t; idx < Nn * din; idx += 256) {
    int n = idx / din, d = idx - n * din;
    const float* row = base + (size_t)n * in_stride;
    float v;
    if (n == 0 || d >= fd) v = row[d];
    else {
      float a = 0.f;
      #pragma unroll
      for (int c = 0; c < NWAY; c++) a += row[fd + c] * anc[c * 256 + d];
      v = 0.5f * a + 0.5f * row[d];
    }
    ob[(size_t)n * GSTR + d] = v;
  }
}

// ---------------------------------------------------------------------------
// score: s[m] = leaky(bn(h4[m,:])) . out_W + out_b   (64 -> 1)
// ---------------------------------------------------------------------------
__global__ __launch_bounds__(256)
void score_kernel(const _Float16* __restrict__ H, const float* __restrict__ fin,
                  const float* __restrict__ gamma, const float* __restrict__ beta,
                  const float* __restrict__ outW, const float* __restrict__ outb,
                  float* __restrict__ S)
{
  __shared__ float sp[320]; // mean | rstd | gamma | beta | outW
  int t = threadIdx.x;
  if (t < 64) {
    sp[t]       = fin[t];
    sp[64 + t]  = fin[64 + t];
    sp[128 + t] = gamma[t];
    sp[192 + t] = beta[t];
    sp[256 + t] = outW[t];
  }
  __syncthreads();
  int m = blockIdx.x * 256 + t;
  const h8* hp = (const h8*)(H + (size_t)m * 64);
  float acc = outb[0];
  #pragma unroll
  for (int cidx = 0; cidx < 8; cidx++) {
    h8 hv = hp[cidx];
    #pragma unroll
    for (int e = 0; e < 8; e++) {
      int k = cidx * 8 + e;
      float f = (float)hv[e];
      f = leaky((f - sp[k]) * sp[64 + k] * sp[128 + k] + sp[192 + k]);
      acc += f * sp[256 + k];
    }
  }
  S[m] = acc;
}

// ---------------------------------------------------------------------------
// softmax + GCN:  p = softmax(s_row - 1e8*diag);  u = g_i - sum_j p_j g_j;
// g_new = (leaky)(W u + b);  write concat(g_new, g) or final output row 0.
// One block per (b, i) node row.
// ---------------------------------------------------------------------------
__global__ __launch_bounds__(384)
void softmax_gcn_kernel(const float* __restrict__ S, const float* __restrict__ GA,
                        int din, const float* __restrict__ Wg,
                        const float* __restrict__ bg, int dout, int is_final,
                        float* __restrict__ Gn, float* __restrict__ out)
{
  __shared__ float p[Nn];
  __shared__ float u[264];
  __shared__ float red[Nn];
  int t  = threadIdx.x;
  int bi = blockIdx.x;
  int b  = bi / Nn;
  int i  = bi - b * Nn;

  float v = S[(size_t)bi * Nn + t];
  if (t == i) v -= 1.0e8f;
  red[t] = v;
  __syncthreads();
  if (t < 128) red[t] = fmaxf(red[t], fmaxf(red[t + 128], red[t + 256]));
  __syncthreads();
  for (int s = 64; s > 0; s >>= 1) { if (t < s) red[t] = fmaxf(red[t], red[t + s]); __syncthreads(); }
  float mx = red[0];
  __syncthreads();
  float e = __expf(v - mx);
  red[t] = e;
  __syncthreads();
  if (t < 128) red[t] = red[t] + red[t + 128] + red[t + 256];
  __syncthreads();
  for (int s = 64; s > 0; s >>= 1) { if (t < s) red[t] += red[t + s]; __syncthreads(); }
  float inv = 1.f / red[0];
  p[t] = e * inv;
  __syncthreads();

  const float* gb = GA + (size_t)b * Nn * GSTR;
  for (int d = t; d < din; d += 384) {
    float acc = 0.f;
    for (int j = 0; j < Nn; j++) acc += p[j] * gb[(size_t)j * GSTR + d];
    u[d] = gb[(size_t)i * GSTR + d] - acc;
  }
  __syncthreads();

  float gn = 0.f;
  if (t < dout) {
    float acc = bg[t];
    const float* wr = Wg + (size_t)t * din;
    for (int d = 0; d < din; d++) acc += wr[d] * u[d];
    gn = is_final ? acc : leaky(acc);
  }
  if (is_final) {
    if (i == 0 && t < dout) out[b * dout + t] = gn;
  } else {
    float* orow = Gn + (size_t)bi * GSTR;
    if (t < dout) orow[t] = gn;
    for (int d = t; d < din; d += 384) orow[dout + d] = gb[(size_t)i * GSTR + d];
  }
}

// ---------------------------------------------------------------------------
extern "C" void kernel_launch(void* const* d_in, const int* in_sizes, int n_in,
                              void* d_out, int out_size, void* d_ws, size_t ws_size,
                              hipStream_t stream)
{
  (void)out_size; (void)ws_size;
  // --- input parsing: 'x' first (dict order), params in jax sorted-key order:
  //     final{dist{convs[W,b,be,g]x4, out_W, out_b}, gcn{W,b}}, layers[0], layers[1]
  int pi = 0;
  const float* x;
  if (in_sizes[0] == Bb * Nn * IND) { x = (const float*)d_in[0]; pi = 1; }
  else                              { x = (const float*)d_in[n_in - 1]; pi = 0; }

  struct DistP { const float* W[4]; const float* ga[4]; const float* be[4];
                 const float* outW; const float* outb; };
  struct GcnP  { const float* W; const float* b; };
  auto parseDist = [&](DistP& dp) {
    for (int c = 0; c < 4; c++) {
      dp.W[c]  = (const float*)d_in[pi++];  // 'W'
      pi++;                                 // 'b'  (cancels through BN)
      dp.be[c] = (const float*)d_in[pi++];  // 'be' (beta)
      dp.ga[c] = (const float*)d_in[pi++];  // 'g'  (gamma)
    }
    dp.outW = (const float*)d_in[pi++];
    dp.outb = (const float*)d_in[pi++];
  };
  auto parseGcn = [&](GcnP& gp) {
    gp.W = (const float*)d_in[pi++];
    gp.b = (const float*)d_in[pi++];
  };

  DistP fd, l0d, l1d; GcnP fg, l0g, l1g;
  parseDist(fd);  parseGcn(fg);    // 'final' < 'layers'
  parseDist(l0d); parseGcn(l0g);
  parseDist(l1d); parseGcn(l1g);

  // --- workspace carve ---
  char* wp = (char*)d_ws;
  auto take = [&](size_t bytes) -> void* {
    void* r = wp; wp += (bytes + 255) & ~(size_t)255; return r;
  };
  float*    GA  = (float*)take((size_t)Bb * Nn * GSTR * 4);   // anchored g
  float*    GN  = (float*)take((size_t)Bb * Nn * GSTR * 4);   // concat(g_new, g)
  _Float16* HX  = (_Float16*)take((size_t)Mtot * 128 * 2);    // ping
  _Float16* HY  = (_Float16*)take((size_t)Mtot * 128 * 2);    // pong
  float*    S   = (float*)take((size_t)Mtot * 4);             // pair scores
  float*    raw = (float*)take(1024);                         // BN sum/sumsq
  float*    fin = (float*)take(1024);                         // BN mean/rstd
  float*    outp = (float*)d_out;

  const int MB = Mtot / 64;
  const dim3 gblk(128);

  auto runLayer = [&](const float* gin, int istr, int din, const DistP& dp,
                      const GcnP& gp, int gdout, int isfinal, float* gnext) {
    anchor_kernel<<<dim3(Bb), dim3(256), 0, stream>>>(gin, istr, din, GA);
    zero_stats<<<dim3(1), dim3(256), 0, stream>>>(raw);
    // conv0: |g_i - g_j| (din) -> 128   (full 128-wide N tile: A staged once)
    dist_gemm<0, 128><<<dim3(MB, 1), gblk, 0, stream>>>(GA, (const _Float16*)nullptr, din,
                                                        dp.W[0], nullptr, nullptr, nullptr,
                                                        HX, 128, raw);
    finalize_stats<<<dim3(1), dim3(256), 0, stream>>>(raw, fin, 128);
    // conv1: 128 -> 128 (BN/leaky of conv0 applied in A-load)
    dist_gemm<1, 128><<<dim3(MB, 1), gblk, 0, stream>>>(nullptr, HX, 128, dp.W[1],
                                                        fin, dp.ga[0], dp.be[0], HY, 128, raw);
    finalize_stats<<<dim3(1), dim3(256), 0, stream>>>(raw, fin, 128);
    // conv2: 128 -> 64
    dist_gemm<1, 64><<<dim3(MB, 1), gblk, 0, stream>>>(nullptr, HY, 128, dp.W[2],
                                                       fin, dp.ga[1], dp.be[1], HX, 64, raw);
    finalize_stats<<<dim3(1), dim3(256), 0, stream>>>(raw, fin, 64);
    // conv3: 64 -> 64
    dist_gemm<1, 64><<<dim3(MB, 1), gblk, 0, stream>>>(nullptr, HX, 64, dp.W[3],
                                                       fin, dp.ga[2], dp.be[2], HY, 64, raw);
    finalize_stats<<<dim3(1), dim3(256), 0, stream>>>(raw, fin, 64);
    // out projection 64 -> 1
    score_kernel<<<dim3(Mtot / 256), dim3(256), 0, stream>>>(HY, fin, dp.ga[3], dp.be[3],
                                                             dp.outW, dp.outb, S);
    // softmax + gcn (+concat or final output)
    softmax_gcn_kernel<<<dim3(Bb * Nn), dim3(384), 0, stream>>>(S, GA, din, gp.W, gp.b,
                                                                gdout, isfinal, gnext, outp);
  };

  runLayer(x,  IND,  IND,       l0d, l0g, 64,   0, GN);
  runLayer(GN, GSTR, IND + 64,  l1d, l1g, 64,   0, GN);
  runLayer(GN, GSTR, IND + 128, fd,  fg,  NWAY, 1, nullptr);
}
